// ResidualSparseConvBackbone_59768764891690
// MI455X (gfx1250) — compile-verified, matrix operand order checked by reading, not run on hardware
//
#include <hip/hip_runtime.h>
#include <hip/hip_bf16.h>

typedef __attribute__((ext_vector_type(16))) _Float16 v16h;
typedef __attribute__((ext_vector_type(8)))  _Float16 v8h;
typedef __attribute__((ext_vector_type(8)))  float    v8f;

#define EPSV 1e-5f

// ---------------------------------------------------------------------------
// utility: zero fill
// ---------------------------------------------------------------------------
__global__ __launch_bounds__(256) void fill_zero(float* __restrict__ p, long n) {
    long i = (long)blockIdx.x * blockDim.x + threadIdx.x;
    if (i < n) p[i] = 0.0f;
}

// ---------------------------------------------------------------------------
// fp32 -> fp16 elementwise conversion (activations, once per conv input).
// ---------------------------------------------------------------------------
__global__ __launch_bounds__(256)
void cvt_f16(const float* __restrict__ X, _Float16* __restrict__ Xh, long n) {
    long i = (long)blockIdx.x * blockDim.x + threadIdx.x;
    if (i < n) Xh[i] = (_Float16)X[i];
}

// ---------------------------------------------------------------------------
// Repack weights [K, Cin, Cout] (f32 row-major) into WMMA B-fragment order:
//   Wh[(((k*nc + cbi)*nt + n_t)*32 + lane)*16 + e]
// element e -> K index kk = cbi*32 + khalf + ((e&8)?16:0) + (e&7),
// khalf = (lane>=16)?8:0, column = n_t*16 + (lane&15). Zero-pads kk >= Cin.
// ---------------------------------------------------------------------------
__global__ __launch_bounds__(256)
void repack_w(const float* __restrict__ W, _Float16* __restrict__ Wh,
              int Cin, int Cout, int K, int nc, int nt)
{
    long i = (long)blockIdx.x * blockDim.x + threadIdx.x;
    long total = (long)K * nc * nt * 512;
    if (i >= total) return;
    int  e    = (int)(i & 15);
    int  lane = (int)((i >> 4) & 31);
    long t    = i >> 9;
    int  n_t  = (int)(t % nt);
    int  cbi  = (int)((t / nt) % nc);
    int  k    = (int)(t / ((long)nt * nc));
    int  khalf = (lane >> 4) << 3;
    int  kk   = cbi * 32 + khalf + ((e & 8) ? 16 : 0) + (e & 7);
    int  col  = n_t * 16 + (lane & 15);
    float v = (kk < Cin) ? W[((long)k * Cin + kk) * Cout + col] : 0.0f;
    Wh[i] = (_Float16)v;
}

// ---------------------------------------------------------------------------
// Gather -> WMMA -> scatter-add sparse conv (fp16 in, fp32 accumulate).
// One wave computes a 16(M) x 64(Cout) tile for one kernel offset k:
// the A fragment (gathered rows) is loaded ONCE per 32-chunk and reused by
// 4 WMMAs against 4 pre-repacked B fragments -> 4x less gather traffic.
// ---------------------------------------------------------------------------
__global__ __launch_bounds__(256)
void sparse_conv_wmma(const _Float16* __restrict__ Xh, const _Float16* __restrict__ Wh,
                      const int* __restrict__ idx_in, const int* __restrict__ idx_out,
                      float* __restrict__ Out,
                      int M, int Cin, int Cout, int K, int mt, int ntq)
{
    const int wave = threadIdx.x >> 5;
    const int lane = threadIdx.x & 31;
    long tile = (long)blockIdx.x * 8 + wave;
    long total = (long)K * mt * ntq;
    if (tile >= total) return;

    const int quad = (int)(tile % ntq);               // group of 4 n-tiles
    const int m_t  = (int)((tile / ntq) % mt);
    const int k    = (int)(tile / ((long)ntq * mt));

    const int row16 = lane & 15;
    const int khalf = (lane >> 4) << 3;               // 0 or 8
    const int m0 = m_t * 16;
    const int nt = ntq * 4;                           // Cout/16
    const int nc = (Cin + 31) >> 5;
    const bool vec = ((Cin & 31) == 0);

    const int m_a = m0 + row16;
    const _Float16* xrow = nullptr;
    if (m_a < M) {
        int src = idx_in[(long)k * M + m_a];
        xrow = Xh + (long)src * Cin;
    }
    // B fragment base for n_t = quad*4 (+q), advancing nt*512 halves per cbi
    const _Float16* wfrag = Wh + ((((long)k * nc) * nt + quad * 4) * 32 + lane) * 16;
    const long wstep = (long)nt * 512;

    v8f acc0 = {}, acc1 = {}, acc2 = {}, acc3 = {};
    for (int cbi = 0; cbi < nc; ++cbi) {
        const int cb = cbi * 32;
        v16h a;
        if (xrow) {
            if (vec) {
                v8h lo = *(const v8h*)(xrow + cb + khalf);
                v8h hi = *(const v8h*)(xrow + cb + khalf + 16);
                #pragma unroll
                for (int e = 0; e < 8; ++e) { a[e] = lo[e]; a[e + 8] = hi[e]; }
            } else {
                #pragma unroll
                for (int e = 0; e < 16; ++e) {
                    int col = cb + khalf + ((e & 8) ? 16 : 0) + (e & 7);
                    a[e] = (col < Cin) ? xrow[col] : (_Float16)0.0f;
                }
            }
        } else {
            #pragma unroll
            for (int e = 0; e < 16; ++e) a[e] = (_Float16)0.0f;
        }
        const _Float16* wb = wfrag + (long)cbi * wstep;
        v16h b0 = *(const v16h*)(wb);
        v16h b1 = *(const v16h*)(wb + 512);
        v16h b2 = *(const v16h*)(wb + 1024);
        v16h b3 = *(const v16h*)(wb + 1536);
        acc0 = __builtin_amdgcn_wmma_f32_16x16x32_f16(false, a, false, b0, (short)0, acc0, false, false);
        acc1 = __builtin_amdgcn_wmma_f32_16x16x32_f16(false, a, false, b1, (short)0, acc1, false, false);
        acc2 = __builtin_amdgcn_wmma_f32_16x16x32_f16(false, a, false, b2, (short)0, acc2, false, false);
        acc3 = __builtin_amdgcn_wmma_f32_16x16x32_f16(false, a, false, b3, (short)0, acc3, false, false);
    }

    // scatter-add: idx_out rows loaded once, reused for all 4 column tiles
    const int col   = lane & 15;
    const int rbase = (lane >> 4) << 3;
    const int n0    = quad * 64;
    int orow[8];
    #pragma unroll
    for (int v = 0; v < 8; ++v) {
        int mm = m0 + rbase + v;
        orow[v] = (mm < M) ? idx_out[(long)k * M + mm] : -1;
    }
    #pragma unroll
    for (int v = 0; v < 8; ++v) {
        if (orow[v] >= 0) {
            float* po = Out + (long)orow[v] * Cout + n0 + col;
            unsafeAtomicAdd(po,      acc0[v]);
            unsafeAtomicAdd(po + 16, acc1[v]);
            unsafeAtomicAdd(po + 32, acc2[v]);
            unsafeAtomicAdd(po + 48, acc3[v]);
        }
    }
}

// ---------------------------------------------------------------------------
// Dense GEMM, two-source concatenated A (fp32, converted in-register) with
// pre-repacked fp16 B fragments; 16 x 64 tile per wave (shared A fragment).
// Implements cat([A0,A1],1) @ W and the plain final matmul (split == Cin).
// N mult of 16; Cout mult of 64; Cin, split mult of 32.
// ---------------------------------------------------------------------------
__global__ __launch_bounds__(256)
void dense_gemm_wmma(const float* __restrict__ A0, const float* __restrict__ A1,
                     int split, int Cin, const _Float16* __restrict__ Wh,
                     float* __restrict__ Out, int N, int Cout, int mt, int ntq)
{
    const int wave = threadIdx.x >> 5;
    const int lane = threadIdx.x & 31;
    long tile = (long)blockIdx.x * 8 + wave;
    long total = (long)mt * ntq;
    if (tile >= total) return;

    const int quad = (int)(tile % ntq);
    const int m_t  = (int)(tile / ntq);
    const int row16 = lane & 15;
    const int khalf = (lane >> 4) << 3;
    const int m0 = m_t * 16;
    const int m_a = m0 + row16;
    const int s1 = Cin - split;
    const int nc = Cin >> 5;
    const int nt = ntq * 4;

    const _Float16* wfrag = Wh + (((long)quad * 4) * 32 + lane) * 16;
    const long wstep = (long)nt * 512;

    v8f acc0 = {}, acc1 = {}, acc2 = {}, acc3 = {};
    for (int cbi = 0; cbi < nc; ++cbi) {
        const int cb = cbi * 32;
        float af[16];
        #pragma unroll
        for (int run = 0; run < 2; ++run) {
            int base = cb + khalf + run * 16;
            const float* p;
            if (base < split) p = A0 + (long)m_a * split + base;
            else              p = A1 + (long)m_a * s1 + (base - split);
            float4 f0 = *(const float4*)(p);
            float4 f1 = *(const float4*)(p + 4);
            af[run*8+0]=f0.x; af[run*8+1]=f0.y; af[run*8+2]=f0.z; af[run*8+3]=f0.w;
            af[run*8+4]=f1.x; af[run*8+5]=f1.y; af[run*8+6]=f1.z; af[run*8+7]=f1.w;
        }
        v16h a;
        #pragma unroll
        for (int e = 0; e < 16; ++e) a[e] = (_Float16)af[e];
        const _Float16* wb = wfrag + (long)cbi * wstep;
        v16h b0 = *(const v16h*)(wb);
        v16h b1 = *(const v16h*)(wb + 512);
        v16h b2 = *(const v16h*)(wb + 1024);
        v16h b3 = *(const v16h*)(wb + 1536);
        acc0 = __builtin_amdgcn_wmma_f32_16x16x32_f16(false, a, false, b0, (short)0, acc0, false, false);
        acc1 = __builtin_amdgcn_wmma_f32_16x16x32_f16(false, a, false, b1, (short)0, acc1, false, false);
        acc2 = __builtin_amdgcn_wmma_f32_16x16x32_f16(false, a, false, b2, (short)0, acc2, false, false);
        acc3 = __builtin_amdgcn_wmma_f32_16x16x32_f16(false, a, false, b3, (short)0, acc3, false, false);
    }

    const int col   = lane & 15;
    const int rbase = (lane >> 4) << 3;
    const int n0    = quad * 64;
    #pragma unroll
    for (int v = 0; v < 8; ++v) {
        float* po = Out + (long)(m0 + rbase + v) * Cout + n0 + col;
        po[0]  = acc0[v];
        po[16] = acc1[v];
        po[32] = acc2[v];
        po[48] = acc3[v];
    }
}

// ---------------------------------------------------------------------------
// BatchNorm statistics: one block per channel, tree reduce; biased variance.
// ---------------------------------------------------------------------------
__global__ __launch_bounds__(256)
void bn_stats(const float* __restrict__ X, int N, int C,
              float* __restrict__ mean, float* __restrict__ var)
{
    const int c = blockIdx.x;
    float s1 = 0.0f, s2 = 0.0f;
    for (int n = threadIdx.x; n < N; n += 256) {
        float v = X[(long)n * C + c];
        s1 += v; s2 += v * v;
    }
    __shared__ float sh1[256], sh2[256];
    sh1[threadIdx.x] = s1; sh2[threadIdx.x] = s2;
    __syncthreads();
    for (int s = 128; s > 0; s >>= 1) {
        if (threadIdx.x < s) {
            sh1[threadIdx.x] += sh1[threadIdx.x + s];
            sh2[threadIdx.x] += sh2[threadIdx.x + s];
        }
        __syncthreads();
    }
    if (threadIdx.x == 0) {
        float m = sh1[0] / (float)N;
        mean[c] = m;
        var[c]  = sh2[0] / (float)N - m * m;
    }
}

// ---------------------------------------------------------------------------
// y = g*(x-mean)*rsqrt(var+eps)+b  [+ residual] [relu]; in-place on X.
// ---------------------------------------------------------------------------
__global__ __launch_bounds__(256)
void bn_apply(float* __restrict__ X, const float* __restrict__ res,
              const float* __restrict__ g, const float* __restrict__ b,
              const float* __restrict__ mean, const float* __restrict__ var,
              long total, int C, int relu)
{
    long i = (long)blockIdx.x * blockDim.x + threadIdx.x;
    if (i >= total) return;
    int c = (int)(i % C);
    float y = g[c] * (X[i] - mean[c]) * rsqrtf(var[c] + EPSV) + b[c];
    if (res)  y += res[i];
    if (relu) y = fmaxf(y, 0.0f);
    X[i] = y;
}

// ---------------------------------------------------------------------------
// Host orchestration
// ---------------------------------------------------------------------------
extern "C" void kernel_launch(void* const* d_in, const int* in_sizes, int n_in,
                              void* d_out, int out_size, void* d_ws, size_t ws_size,
                              hipStream_t stream)
{
    (void)in_sizes; (void)n_in; (void)out_size; (void)ws_size;

    const int N0 = 80000, N1 = 36000, N2 = 16000, N3 = 7000;
    const int C0 = 64, C1 = 128, C2 = 256, C3 = 512;
    const int K = 27;

    auto pf = [&](int i) { return (const float*)d_in[i]; };
    auto pi = [&](int i) { return (const int*)d_in[i]; };

    const float* x = pf(0);
    const int *l0i = pi(98),  *l0o = pi(99);
    const int *l1i = pi(100), *l1o = pi(101);
    const int *l2i = pi(102), *l2o = pi(103);
    const int *l3i = pi(104), *l3o = pi(105);
    const int *d01i = pi(106), *d01o = pi(107);
    const int *d12i = pi(108), *d12o = pi(109);
    const int *d23i = pi(110), *d23o = pi(111);

    // workspace layout (float-sized units)
    float* ws = (float*)d_ws;
    const long S0 = (long)N0 * C0;          // 5,120,000
    const long S1 = (long)N1 * C1;          // 4,608,000
    const long S2 = (long)N2 * C2;          // 4,096,000
    const long SM = S0;                     // max activation slab
    float* E0 = ws;
    float* E1 = E0 + S0;
    float* E2 = E1 + S1;
    float* TA = E2 + S2;
    float* TB = TA + SM;
    float* TC = TB + SM;
    float* stats = TC + SM;                            // mean[512] | var[512]
    _Float16* XH = (_Float16*)(stats + 1024);          // fp16 activations, SM halves
    _Float16* WH = (_Float16*)(stats + 1024 + SM / 2); // fp16 fragment weights
    // WH capacity needed: 27*512*512 halves = 7,077,888 (14.2 MB)

    auto zero = [&](float* p, long n) {
        fill_zero<<<(int)((n + 255) / 256), 256, 0, stream>>>(p, n);
    };
    auto conv = [&](const float* Xp, long Nin, const float* Wp,
                    const int* ii, const int* io,
                    float* Op, int M, int Cin, int Cout, int Nout) {
        int nc = (Cin + 31) >> 5, nt = Cout / 16, ntq = Cout / 64;
        int mt = (M + 15) / 16;
        long nx = Nin * Cin;
        cvt_f16<<<(int)((nx + 255) / 256), 256, 0, stream>>>(Xp, XH, nx);
        long nw = (long)K * nc * nt * 512;
        repack_w<<<(int)((nw + 255) / 256), 256, 0, stream>>>(Wp, WH, Cin, Cout, K, nc, nt);
        zero(Op, (long)Nout * Cout);
        long tiles = (long)K * mt * ntq;
        sparse_conv_wmma<<<(int)((tiles + 7) / 8), 256, 0, stream>>>(
            XH, WH, ii, io, Op, M, Cin, Cout, K, mt, ntq);
    };
    auto bn = [&](float* Xp, const float* resp, const float* g, const float* b,
                  int Nn, int C, bool relu) {
        bn_stats<<<C, 256, 0, stream>>>(Xp, Nn, C, stats, stats + 512);
        long total = (long)Nn * C;
        bn_apply<<<(int)((total + 255) / 256), 256, 0, stream>>>(
            Xp, resp, g, b, stats, stats + 512, total, C, relu ? 1 : 0);
    };
    auto gemm = [&](const float* A0, const float* A1, int split, int Cin,
                    const float* Wp, float* Op, int Nn, int Cout) {
        int nc = Cin >> 5, nt = Cout / 16, ntq = Cout / 64, mt = Nn / 16;
        long nw = (long)nc * nt * 512;
        repack_w<<<(int)((nw + 255) / 256), 256, 0, stream>>>(Wp, WH, Cin, Cout, 1, nc, nt);
        long tiles = (long)mt * ntq;
        dense_gemm_wmma<<<(int)((tiles + 7) / 8), 256, 0, stream>>>(
            A0, A1, split, Cin, WH, Op, Nn, Cout, mt, ntq);
    };
    // residual block: t2 = relu(bn2(conv2(relu(bn1(conv1(xin))))) + xin)
    auto res_block = [&](float* xin, float* t1, float* t2, int base,
                         const int* ii, const int* io, int M, int Nn, int C) {
        conv(xin, Nn, pf(base + 0), ii, io, t1, M, C, C, Nn);
        bn(t1, nullptr, pf(base + 1), pf(base + 2), Nn, C, true);
        conv(t1, Nn, pf(base + 3), ii, io, t2, M, C, C, Nn);
        bn(t2, xin, pf(base + 4), pf(base + 5), Nn, C, true);
    };

    // ---- stem: e0 = relu(bn(conv(x)))  -> E0
    conv(x, N0, pf(1), l0i, l0o, E0, N0 / 2, 4, C0, N0);
    bn(E0, nullptr, pf(2), pf(3), N0, C0, true);

    // ---- enc1
    conv(E0, N0, pf(4), d01i, d01o, E1, N1, C0, C1, N1);
    bn(E1, nullptr, pf(5), pf(6), N1, C1, true);
    res_block(E1, TA, TB, 7,  l1i, l1o, N1 / 2, N1, C1);   // -> TB
    res_block(TB, TA, E1, 13, l1i, l1o, N1 / 2, N1, C1);   // -> E1 (skip)

    // ---- enc2
    conv(E1, N1, pf(19), d12i, d12o, E2, N2, C1, C2, N2);
    bn(E2, nullptr, pf(20), pf(21), N2, C2, true);
    res_block(E2, TA, TB, 22, l2i, l2o, N2 / 2, N2, C2);   // -> TB
    res_block(TB, TA, E2, 28, l2i, l2o, N2 / 2, N2, C2);   // -> E2 (skip)

    // ---- enc3
    conv(E2, N2, pf(34), d23i, d23o, TA, N3, C2, C3, N3);
    bn(TA, nullptr, pf(35), pf(36), N3, C3, true);
    res_block(TA, TB, TC, 37, l3i, l3o, N3 / 2, N3, C3);   // -> TC
    res_block(TC, TA, TB, 43, l3i, l3o, N3 / 2, N3, C3);   // e3 -> TB

    // ---- dec3: up-conv (swapped maps), cat-gemm with e2, 2x res
    conv(TB, N3, pf(49), d23o, d23i, TA, N3, C3, C2, N2);
    bn(TA, nullptr, pf(50), pf(51), N2, C2, true);
    gemm(TA, E2, C2, 2 * C2, pf(52), TC, N2, C2);          // d2 -> TC
    res_block(TC, TA, TB, 53, l2i, l2o, N2 / 2, N2, C2);   // -> TB
    res_block(TB, TA, TC, 59, l2i, l2o, N2 / 2, N2, C2);   // -> TC

    // ---- dec2
    conv(TC, N2, pf(65), d12o, d12i, TA, N2, C2, C1, N1);
    bn(TA, nullptr, pf(66), pf(67), N1, C1, true);
    gemm(TA, E1, C1, 2 * C1, pf(68), TB, N1, C1);          // d1 -> TB
    res_block(TB, TC, TA, 69, l1i, l1o, N1 / 2, N1, C1);   // -> TA
    res_block(TA, TC, TB, 75, l1i, l1o, N1 / 2, N1, C1);   // -> TB

    // ---- dec1
    conv(TB, N1, pf(81), d01o, d01i, TA, N1, C1, C0, N0);
    bn(TA, nullptr, pf(82), pf(83), N0, C0, true);
    gemm(TA, E0, C0, 2 * C0, pf(84), TC, N0, C0);          // d0 -> TC
    res_block(TC, TA, TB, 85, l0i, l0o, N0 / 2, N0, C0);   // -> TB
    res_block(TB, TA, TC, 91, l0i, l0o, N0 / 2, N0, C0);   // -> TC

    // ---- final: out = d0 @ final_w
    gemm(TC, TC, C0, C0, pf(97), (float*)d_out, N0, 64);
}
